// DotProductAttention_36309653520979
// MI455X (gfx1250) — matrix operand order repaired and checked by reading, not compiled, not run on hardware
//
#include <hip/hip_runtime.h>
#include <hip/hip_bf16.h>
#include <math.h>

// ---------------------------------------------------------------------------
// DotProductAttention (no softmax)  ==>  pure linear chain:
//   qp = q @ Wq^T + bq ; k = x @ Wk^T + bk ; v = x @ Wv^T + bv
//   out = qp @ ((k^T @ v) / sqrt(H))          (exact reassociation:
//   O(B*H^2*L) instead of O(B*L^2*H), no 268MB attn intermediate)
// All GEMMs on v_wmma_f32_16x16x32_bf16, fp32 accumulation, double-buffered
// LDS with one barrier per K-chunk.
// ---------------------------------------------------------------------------

typedef __attribute__((ext_vector_type(16))) __bf16        v16bf;
typedef __attribute__((ext_vector_type(4)))  __bf16        v4bf;
typedef __attribute__((ext_vector_type(8)))  float         v8f;
typedef __attribute__((ext_vector_type(4)))  unsigned int  v4u;
typedef __attribute__((ext_vector_type(2)))  unsigned int  v2u;

#define BM      128
#define BN      128
#define BK      32
#define LDSK    40      // BK + 8 halves of padding -> 80-byte row stride (16B aligned)
#define THREADS 256     // 8 wave32 waves

__device__ __forceinline__ unsigned short bfbits(float f) {
    __bf16 h = (__bf16)f;                       // backend-native cvt (RNE)
    return __builtin_bit_cast(unsigned short, h);
}

// native global-load vector type per operand element type
template <typename TIN> struct LV;
template <> struct LV<float>          { using T = float4; };
template <> struct LV<unsigned short> { using T = v2u;    };

// 4 elements -> native bf16x4 vector (lowered to 2x v_cvt_pk_bf16_f32)
__device__ __forceinline__ v4bf to_bfv(const float4& v) {
    v4bf r;
    r.x = (__bf16)v.x; r.y = (__bf16)v.y; r.z = (__bf16)v.z; r.w = (__bf16)v.w;
    return r;
}
__device__ __forceinline__ v4bf to_bfv(const v2u& v) {   // already bf16: raw copy
    return __builtin_bit_cast(v4bf, v);
}

// 4 elements -> 4 raw bf16 scalars (transpose staging, scalar ds_store_b16)
__device__ __forceinline__ void to_bf4(const float4& v, unsigned short s[4]) {
    s[0] = bfbits(v.x); s[1] = bfbits(v.y); s[2] = bfbits(v.z); s[3] = bfbits(v.w);
}
__device__ __forceinline__ void to_bf4(const v2u& v, unsigned short s[4]) {
    s[0] = (unsigned short)(v.x & 0xFFFFu); s[1] = (unsigned short)(v.x >> 16);
    s[2] = (unsigned short)(v.y & 0xFFFFu); s[3] = (unsigned short)(v.y >> 16);
}

// OPA==0: A is [M][K] row-major.  OPA==1: A is [K][M] (we compute with A^T).
// OPB==0: B is [K][N] row-major.  OPB==1: B is [N][K] (torch weight layout, B^T).
// TIN:    fp32 or raw-bf16 (unsigned short) global operands.
// OUTBF:  store C as raw bf16 (intermediates) or fp32 (final output).
// All of M,N are multiples of 128 and K of 32 for this problem -> no edge guards.
template <int OPA, int OPB, bool BIAS, typename TIN, bool OUTBF>
__global__ __launch_bounds__(THREADS)
void wmma_gemm(const TIN* __restrict__ Ag, const TIN* __restrict__ Bg,
               const float* __restrict__ bias, void* __restrict__ Cg,
               int M, int N, int K, int lda, int ldb, int ldc,
               long long sA, long long sB, long long sC, float scale)
{
    using LT = typename LV<TIN>::T;

    __shared__ __align__(16) unsigned short As[2][BM * LDSK]; // As[b][m][kk]
    __shared__ __align__(16) unsigned short Bs[2][BN * LDSK]; // Bs[b][n][kk] (K-contig)

    const int t     = threadIdx.x;
    const int lane  = t & 31;
    const int wave  = t >> 5;
    const int waveM = wave & 3;   // 4 waves along M: 32 rows each
    const int waveN = wave >> 2;  // 2 waves along N: 64 cols each

    const int m0 = blockIdx.y * BM;
    const int n0 = blockIdx.x * BN;
    const int b  = blockIdx.z;

    const TIN* A = Ag + (size_t)b * (size_t)sA;
    const TIN* B = Bg + (size_t)b * (size_t)sB;

    const int halfSel = (lane < 16) ? 0 : 1;  // which K-half this lane owns
    const int mn      = lane & 15;            // row (A) / col (B,D) within 16x16 tile

    // ---- per-thread staging indices (constant across K loop)
    const int drR = t >> 3, drC = (t & 7) * 4;    // direct path: 32 rows/pass, 8x4 K
    const int trK = t >> 5, trC = (t & 31) * 4;   // transpose path: 8 k/pass, 32x4 cols

    // ---- per-wave LDS fragment offsets (ushort indices)
    const int aoff = (waveM * 32 + mn) * LDSK + halfSel * 8;   // + mi*16*LDSK
    const int boff = (waveN * 64 + mn) * LDSK + halfSel * 8;   // + ni*16*LDSK

    v8f acc[2][4];
    {
        const v8f z = {0.f, 0.f, 0.f, 0.f, 0.f, 0.f, 0.f, 0.f};
        #pragma unroll
        for (int i = 0; i < 2; ++i)
            #pragma unroll
            for (int j = 0; j < 4; ++j) acc[i][j] = z;
    }

    LT regsA[4], regsB[4];

    auto loadTiles = [&](int k0) {
        if (OPA == 0) {
            #pragma unroll
            for (int p = 0; p < 4; ++p)
                regsA[p] = *(const LT*)&A[(size_t)(m0 + p * 32 + drR) * lda + (k0 + drC)];
        } else {
            #pragma unroll
            for (int p = 0; p < 4; ++p)
                regsA[p] = *(const LT*)&A[(size_t)(k0 + p * 8 + trK) * lda + (m0 + trC)];
        }
        if (OPB == 1) {
            #pragma unroll
            for (int p = 0; p < 4; ++p)
                regsB[p] = *(const LT*)&B[(size_t)(n0 + p * 32 + drR) * ldb + (k0 + drC)];
        } else {
            #pragma unroll
            for (int p = 0; p < 4; ++p)
                regsB[p] = *(const LT*)&B[(size_t)(k0 + p * 8 + trK) * ldb + (n0 + trC)];
        }
    };

    auto storeTiles = [&](int buf) {
        if (OPA == 0) {
            #pragma unroll
            for (int p = 0; p < 4; ++p)
                *(v4bf*)&As[buf][(p * 32 + drR) * LDSK + drC] = to_bfv(regsA[p]);
        } else {
            #pragma unroll
            for (int p = 0; p < 4; ++p) {
                unsigned short s[4]; to_bf4(regsA[p], s);
                #pragma unroll
                for (int j = 0; j < 4; ++j) As[buf][(trC + j) * LDSK + (p * 8 + trK)] = s[j];
            }
        }
        if (OPB == 1) {
            #pragma unroll
            for (int p = 0; p < 4; ++p)
                *(v4bf*)&Bs[buf][(p * 32 + drR) * LDSK + drC] = to_bfv(regsB[p]);
        } else {
            #pragma unroll
            for (int p = 0; p < 4; ++p) {
                unsigned short s[4]; to_bf4(regsB[p], s);
                #pragma unroll
                for (int j = 0; j < 4; ++j) Bs[buf][(trC + j) * LDSK + (p * 8 + trK)] = s[j];
            }
        }
    };

    // ---- pipeline prologue: tile 0 into buffer 0
    loadTiles(0);
    storeTiles(0);
    __syncthreads();

    int cur = 0;
    for (int k0 = 0; k0 < K; k0 += BK) {
        const bool more = (k0 + BK) < K;

        // issue next tile's global loads early; they retire under the WMMAs
        if (more) loadTiles(k0 + BK);

        // L2 prefetch two tiles ahead (global_prefetch_b8)
        if (k0 + 2 * BK < K) {
            if (OPA == 0) __builtin_prefetch(&A[(size_t)(m0 + drR) * lda + (k0 + 2 * BK + drC)], 0, 1);
            else          __builtin_prefetch(&A[(size_t)(k0 + 2 * BK + trK) * lda + (m0 + trC)], 0, 1);
            if (OPB == 1) __builtin_prefetch(&B[(size_t)(n0 + drR) * ldb + (k0 + 2 * BK + drC)], 0, 1);
            else          __builtin_prefetch(&B[(size_t)(k0 + 2 * BK + trK) * ldb + (n0 + trC)], 0, 1);
        }

        // -------- WMMA fragments straight from LDS (2x ds_load_b128 each)
        union FR { v16bf bf; v4u u[2]; };
        FR afrag[2], bfrag[4];
        #pragma unroll
        for (int mi = 0; mi < 2; ++mi) {
            // A 16x32 bf16 layout: lane<16 -> K 0..7/16..23 ; lane>=16 -> K 8..15/24..31
            const unsigned short* pa = &As[cur][aoff + mi * 16 * LDSK];
            afrag[mi].u[0] = *(const v4u*)(pa);
            afrag[mi].u[1] = *(const v4u*)(pa + 16);
        }
        #pragma unroll
        for (int ni = 0; ni < 4; ++ni) {
            // B 32x16 bf16 layout: lane<16 -> K 0..7/16..23 of col N=lane ; lane>=16 -> K 8..15/24..31
            const unsigned short* pb = &Bs[cur][boff + ni * 16 * LDSK];
            bfrag[ni].u[0] = *(const v4u*)(pb);
            bfrag[ni].u[1] = *(const v4u*)(pb + 16);
        }

        #pragma unroll
        for (int mi = 0; mi < 2; ++mi)
            #pragma unroll
            for (int ni = 0; ni < 4; ++ni)
                acc[mi][ni] = __builtin_amdgcn_wmma_f32_16x16x32_bf16(
                    /*neg_a=*/false, afrag[mi].bf,
                    /*neg_b=*/false, bfrag[ni].bf,
                    /*c_mod=*/(short)0, acc[mi][ni],
                    /*reuse_a=*/false, /*reuse_b=*/false);

        // convert+commit next tile into the other buffer; single barrier per iter
        if (more) {
            storeTiles(cur ^ 1);
            __syncthreads();
        }
        cur ^= 1;
    }

    // -------- epilogue: D layout -> lane: N = lane%16 ; M = (lane<16 ? r : 8+r)
    #pragma unroll
    for (int mi = 0; mi < 2; ++mi) {
        #pragma unroll
        for (int ni = 0; ni < 4; ++ni) {
            const int n = n0 + waveN * 64 + ni * 16 + mn;
            const float badd = BIAS ? bias[n] : 0.0f;
            #pragma unroll
            for (int r = 0; r < 8; ++r) {
                const int m = m0 + waveM * 32 + mi * 16 + halfSel * 8 + r;
                const float val = acc[mi][ni][r] * scale + badd;
                const size_t idx = (size_t)b * (size_t)sC + (size_t)m * ldc + n;
                if (OUTBF) ((unsigned short*)Cg)[idx] = bfbits(val);
                else       ((float*)Cg)[idx]          = val;
            }
        }
    }
}

// ---------------------------------------------------------------------------
// Launch pipeline.
//   B=16, LQ=LKV=2048, DQ=DKV=H=512, fp32 in / fp32 out.
// Workspace (bf16 intermediates): qp 32MB | k 32MB | v 32MB | S 8MB = 104MB.
// ---------------------------------------------------------------------------
extern "C" void kernel_launch(void* const* d_in, const int* in_sizes, int n_in,
                              void* d_out, int out_size, void* d_ws, size_t ws_size,
                              hipStream_t stream)
{
    (void)in_sizes; (void)n_in; (void)out_size; (void)ws_size;

    const float* q  = (const float*)d_in[0];
    const float* x  = (const float*)d_in[1];
    const float* Wq = (const float*)d_in[2];
    const float* bq = (const float*)d_in[3];
    const float* Wk = (const float*)d_in[4];
    const float* bk = (const float*)d_in[5];
    const float* Wv = (const float*)d_in[6];
    const float* bv = (const float*)d_in[7];
    float* out = (float*)d_out;

    const int Bb = 16, LQ = 2048, LKV = 2048, D = 512, H = 512;

    unsigned short* qp = (unsigned short*)d_ws;              // [B*LQ, H]  bf16
    unsigned short* kk = qp + (size_t)Bb * LQ * H;           // [B*LKV,H]  bf16
    unsigned short* vv = kk + (size_t)Bb * LKV * H;          // [B*LKV,H]  bf16
    unsigned short* ss = vv + (size_t)Bb * LKV * H;          // [B, H, H]  bf16

    const dim3 blk(THREADS);

    // ---- projections: [B*L,512] x [512,512]^T (+bias), batch folded into M
    const dim3 gProj(H / BN, (Bb * LQ) / BM, 1);
    wmma_gemm<0, 1, true, float, true><<<gProj, blk, 0, stream>>>(
        q, Wq, bq, qp, Bb * LQ, H, D, D, D, H, 0, 0, 0, 1.0f);
    wmma_gemm<0, 1, true, float, true><<<gProj, blk, 0, stream>>>(
        x, Wk, bk, kk, Bb * LKV, H, D, D, D, H, 0, 0, 0, 1.0f);
    wmma_gemm<0, 1, true, float, true><<<gProj, blk, 0, stream>>>(
        x, Wv, bv, vv, Bb * LKV, H, D, D, D, H, 0, 0, 0, 1.0f);

    // ---- S[b] = (k[b]^T @ v[b]) / sqrt(H)   (512x512, K=2048, per batch)
    const float scale = 0.044194173824159216f; // 1/sqrt(512)
    const dim3 gS(H / BN, H / BM, Bb);
    wmma_gemm<1, 0, false, unsigned short, true><<<gS, blk, 0, stream>>>(
        kk, vv, nullptr, ss, H, H, LKV, H, H, H,
        (long long)LKV * H, (long long)LKV * H, (long long)H * H, scale);

    // ---- out[b] = qp[b] @ S[b]              (2048x512, K=512, per batch, fp32 out)
    const dim3 gO(H / BN, LQ / BM, Bb);
    wmma_gemm<0, 0, false, unsigned short, false><<<gO, blk, 0, stream>>>(
        qp, ss, nullptr, out, LQ, H, H, H, H, H,
        (long long)LQ * H, (long long)H * H, (long long)LQ * H, 1.0f);
}
